// sptialattention_62328565399660
// MI455X (gfx1250) — matrix operand order repaired and checked by reading, not compiled, not run on hardware
//
#include <hip/hip_runtime.h>
#include <hip/hip_bf16.h>
#include <math.h>

// ---------------------------------------------------------------------------
// Spatial attention pipeline for MI455X (gfx1250, wave32, WMMA).
// 4 big GEMMs run as bf16 v_wmma_f32_16x16x32_bf16 with f32 accumulation,
// software-pipelined (register prefetch + L2 prefetch two tiles ahead).
// ---------------------------------------------------------------------------

typedef __attribute__((ext_vector_type(16))) __bf16 v16bf;
typedef __attribute__((ext_vector_type(8)))  __bf16 v8bf;
typedef __attribute__((ext_vector_type(8)))  float  v8f;

#define B_    2048
#define N_    49
#define C_    768
#define NH_   12
#define HD_   64
#define HD2_  32
#define C2_   384
#define TS_   32
#define SCALE_ 0.125f   // HD^-0.5

// ======================================================================
// Generic GEMM:  Y(M,N) = X(M,K) * W(N,K)^T + bias(N)      (all f32 I/O)
// M,N multiples of 128; K multiple of 32.  bf16 WMMA, f32 accum.
// Block: 256 thr = 8 waves; block tile 128x128; wave tile 32x64.
// Pipelined: while WMMAs consume LDS tile k0, tile k0+32 is in flight to
// registers and tile k0+64 is being prefetched into L2.
// ======================================================================
__global__ __launch_bounds__(256)
void gemm_xwt_bias(const float* __restrict__ X, const float* __restrict__ W,
                   const float* __restrict__ bias, float* __restrict__ Y,
                   int M, int N, int K)
{
    __shared__ __bf16 As[128][40];   // X tile [m][k]; 80B row stride keeps 16B alignment
    __shared__ __bf16 Bs[128][40];   // W tile [n][k]

    const int tid   = threadIdx.x;
    const int lane  = tid & 31;
    const int wave  = tid >> 5;
    const int waveM = wave & 3;      // 0..3  -> 32-row slice
    const int waveN = wave >> 2;     // 0..1  -> 64-col slice
    const int lmod  = lane & 15;
    const bool lo16 = lane < 16;

    const int row0 = blockIdx.y * 128;
    const int col0 = blockIdx.x * 128;

    v8f acc[2][4] = {};

    const int ldr = tid >> 3;         // 0..31
    const int ldc = (tid & 7) * 4;    // 0,4,...,28

    // ---- prologue: fetch tile k0=0 into registers ----
    float4 pa[4], pb[4];
    {
        const float* xp = &X[(size_t)(row0 + ldr) * K + ldc];
        const float* wp = &W[(size_t)(col0 + ldr) * K + ldc];
        for (int p = 0; p < 4; ++p) {
            pa[p] = *(const float4*)(xp + (size_t)(p * 32) * K);
            pb[p] = *(const float4*)(wp + (size_t)(p * 32) * K);
        }
    }

    for (int k0 = 0; k0 < K; k0 += 32) {
        // ---- stage prefetched registers into LDS (f32 -> bf16 RNE) ----
        for (int p = 0; p < 4; ++p) {
            int r = p * 32 + ldr;
            As[r][ldc + 0] = (__bf16)pa[p].x; As[r][ldc + 1] = (__bf16)pa[p].y;
            As[r][ldc + 2] = (__bf16)pa[p].z; As[r][ldc + 3] = (__bf16)pa[p].w;
            Bs[r][ldc + 0] = (__bf16)pb[p].x; Bs[r][ldc + 1] = (__bf16)pb[p].y;
            Bs[r][ldc + 2] = (__bf16)pb[p].z; Bs[r][ldc + 3] = (__bf16)pb[p].w;
        }
        __syncthreads();

        // ---- kick off next tile's global loads (overlap with WMMA) ----
        if (k0 + 32 < K) {
            const float* xp = &X[(size_t)(row0 + ldr) * K + (k0 + 32) + ldc];
            const float* wp = &W[(size_t)(col0 + ldr) * K + (k0 + 32) + ldc];
            for (int p = 0; p < 4; ++p) {
                pa[p] = *(const float4*)(xp + (size_t)(p * 32) * K);
                pb[p] = *(const float4*)(wp + (size_t)(p * 32) * K);
            }
        }
        // ---- L2 prefetch two tiles ahead (global_prefetch_b8) ----
        if (k0 + 64 < K) {
            __builtin_prefetch(&X[(size_t)(row0 + ldr) * K + (k0 + 64) + ldc], 0, 1);
            __builtin_prefetch(&W[(size_t)(col0 + ldr) * K + (k0 + 64) + ldc], 0, 1);
        }

        // ---- build fragments per documented CDNA5 layouts ----
        v16bf afr[2], bfr[4];
        {
            // A 16x32 bf16: lane m=lane%16; lo lanes K-runs [0..7],[16..23]; hi [8..15],[24..31]
            const int kb = lo16 ? 0 : 8;
            for (int mi = 0; mi < 2; ++mi) {
                const __bf16* arow = &As[waveM * 32 + mi * 16 + lmod][0];
                v8bf x0 = *(const v8bf*)(arow + kb);
                v8bf x1 = *(const v8bf*)(arow + kb + 16);
                afr[mi] = __builtin_shufflevector(x0, x1,
                          0,1,2,3,4,5,6,7,8,9,10,11,12,13,14,15);
            }
            // B 32x16 bf16: lane n=lane%16; lo lanes K-runs [0..7],[16..23]; hi [8..15],[24..31]
            const int kbb = lo16 ? 0 : 8;
            for (int ni = 0; ni < 4; ++ni) {
                const __bf16* brow = &Bs[waveN * 64 + ni * 16 + lmod][0];
                v8bf x0 = *(const v8bf*)(brow + kbb);
                v8bf x1 = *(const v8bf*)(brow + kbb + 16);
                bfr[ni] = __builtin_shufflevector(x0, x1,
                          0,1,2,3,4,5,6,7,8,9,10,11,12,13,14,15);
            }
        }
        for (int mi = 0; mi < 2; ++mi)
            for (int ni = 0; ni < 4; ++ni)
                acc[mi][ni] = __builtin_amdgcn_wmma_f32_16x16x32_bf16(
                    false, afr[mi], false, bfr[ni],
                    (short)0, acc[mi][ni], false, false);
        __syncthreads();
    }

    // ---- epilogue: D layout  m = r + (lane<16?0:8),  n = lane%16 ----
    for (int mi = 0; mi < 2; ++mi) {
        int m0 = row0 + waveM * 32 + mi * 16 + (lo16 ? 0 : 8);
        for (int ni = 0; ni < 4; ++ni) {
            int n = col0 + waveN * 64 + ni * 16 + lmod;
            float bn = bias[n];
            for (int r = 0; r < 8; ++r)
                Y[(size_t)(m0 + r) * N + n] = acc[mi][ni][r] + bn;
        }
    }
}

// ======================================================================
// Per-(b,h): channel mean/max -> Wc GEMV -> exact GELU -> top-32 channels
// (softmax skipped: monotone, only indices consumed), token-mean top-32.
// Stable tie-break = rank by (value desc, index asc), exactly lax.top_k.
// ======================================================================
__global__ __launch_bounds__(64)
void pool_select_kernel(const float* __restrict__ q,
                        const float* __restrict__ Wc, const float* __restrict__ bc,
                        int* __restrict__ cidx, int* __restrict__ pidx)
{
    const int bh = blockIdx.x;
    const int b  = bh / NH_, h = bh % NH_;
    const int t  = threadIdx.x;                 // 0..63
    const float* qb = q + (size_t)b * N_ * C_ + h * HD_;

    __shared__ float feat[128];
    __shared__ float gsc[64];
    __shared__ float tsc[N_];
    __shared__ int   sel[64];

    // channel stats over tokens
    {
        float m = 0.f, mx = -INFINITY;
        for (int n = 0; n < N_; ++n) {
            float v = qb[(size_t)n * C_ + t];
            m += v; mx = fmaxf(mx, v);
        }
        feat[t]      = m * (1.0f / N_);
        feat[64 + t] = mx;
    }
    __syncthreads();

    // logits + exact GELU
    {
        float y = bc[t];
        const float* w = Wc + t * 128;
        for (int j = 0; j < 128; ++j) y += feat[j] * w[j];
        gsc[t] = 0.5f * y * (1.0f + erff(y * 0.70710678118654752f));
    }
    __syncthreads();

    // top-32 channels (rank < 32), emitted in ascending index order
    {
        float gd = gsc[t];
        int rank = 0;
        for (int j = 0; j < 64; ++j) {
            float gj = gsc[j];
            rank += ((gj > gd) || (gj == gd && j < t)) ? 1 : 0;
        }
        sel[t] = (rank < HD2_) ? 1 : 0;
    }
    __syncthreads();
    if (sel[t]) {
        int pfx = 0;
        for (int j = 0; j < t; ++j) pfx += sel[j];
        cidx[(size_t)bh * HD2_ + pfx] = t;
    }
    __syncthreads();

    // token scores = mean over head dim
    if (t < N_) {
        const float* row = q + (size_t)(b * N_ + t) * C_ + h * HD_;
        float s = 0.f;
        for (int d = 0; d < HD_; ++d) s += row[d];
        tsc[t] = s * (1.0f / HD_);
    }
    __syncthreads();
    if (t < N_) {
        float sd = tsc[t];
        int rank = 0;
        for (int j = 0; j < N_; ++j) {
            float sj = tsc[j];
            rank += ((sj > sd) || (sj == sd && j < t)) ? 1 : 0;
        }
        sel[t] = (rank < TS_) ? 1 : 0;
    }
    __syncthreads();
    if (t < N_ && sel[t]) {
        int pfx = 0;
        for (int j = 0; j < t; ++j) pfx += sel[j];
        pidx[(size_t)bh * TS_ + pfx] = t;
    }
}

// ======================================================================
// Gather k-GEMM input:  qcha1[b,n,h*32+j] = q[b,n,h*64 + cidx[b*12+h][j]]
// ======================================================================
__global__ __launch_bounds__(256)
void gather_qcha1(const float* __restrict__ q, const int* __restrict__ cidx,
                  float* __restrict__ out)
{
    size_t idx = (size_t)blockIdx.x * 256 + threadIdx.x;   // < B*N*C2 exactly
    int    c   = (int)(idx % C2_);
    size_t bn  = idx / C2_;
    int    h   = c / HD2_, j = c % HD2_;
    size_t b   = bn / N_;
    int ch = cidx[(b * NH_ + h) * HD2_ + j];
    out[idx] = q[bn * C_ + h * HD_ + ch];
}

// ======================================================================
// Gather v-GEMM input:  vin[b,t,c] = q[b, p[b*12+c/64][t], c]
// ======================================================================
__global__ __launch_bounds__(256)
void gather_vin(const float* __restrict__ q, const int* __restrict__ pidx,
                float* __restrict__ out)
{
    size_t idx = (size_t)blockIdx.x * 256 + threadIdx.x;   // < B*TS*C exactly
    int    c   = (int)(idx % C_);
    size_t bt  = idx / C_;
    int    t   = (int)(bt % TS_);
    size_t b   = bt / TS_;
    int    h   = c / HD_;
    int n = pidx[(b * NH_ + h) * TS_ + t];
    out[idx] = q[(b * N_ + n) * C_ + c];
}

// ======================================================================
// Per-(b,h) attention: attn = kh(49x32)·qsptT + relpos bias,
// gate = sigmoid(mean_t), softmax over t, o = (a·vh) * gate.
// ======================================================================
__global__ __launch_bounds__(64)
void attn_kernel(const float* __restrict__ q, const float* __restrict__ kbuf,
                 const float* __restrict__ vbuf, const float* __restrict__ btab,
                 const int* __restrict__ cidx, const int* __restrict__ pidx,
                 float* __restrict__ obuf)
{
    const int bh = blockIdx.x;
    const int b  = bh / NH_, h = bh % NH_;
    const int tid = threadIdx.x;

    __shared__ float qs[TS_][HD2_];
    __shared__ float kh[N_][HD2_];
    __shared__ float vh[TS_][HD_];
    __shared__ float at[N_][TS_ + 1];
    __shared__ int   pl[TS_];
    __shared__ int   cl[HD2_];

    if (tid < TS_)  pl[tid] = pidx[(size_t)bh * TS_ + tid];
    if (tid < HD2_) cl[tid] = cidx[(size_t)bh * HD2_ + tid];
    __syncthreads();

    for (int idx = tid; idx < TS_ * HD2_; idx += 64) {           // q-selected tile
        int t = idx / HD2_, j = idx % HD2_;
        qs[t][j] = SCALE_ * q[((size_t)(b * N_) + pl[t]) * C_ + h * HD_ + cl[j]];
    }
    for (int idx = tid; idx < N_ * HD2_; idx += 64) {            // k tile
        int i = idx / HD2_, j = idx % HD2_;
        kh[i][j] = kbuf[((size_t)(b * N_) + i) * C2_ + h * HD2_ + j];
    }
    for (int idx = tid; idx < TS_ * HD_; idx += 64) {            // v tile
        int t = idx / HD_, d = idx % HD_;
        vh[t][d] = vbuf[((size_t)(b * TS_) + t) * C_ + h * HD_ + d];
    }
    __syncthreads();

    // attn + relative position bias (closed-form rpi for 7x7 window)
    for (int idx = tid; idx < N_ * TS_; idx += 64) {
        int i = idx / TS_, t = idx % TS_;
        float s = 0.f;
        for (int j = 0; j < HD2_; ++j) s += kh[i][j] * qs[t][j];
        int n  = pl[t];
        int r0 = (i / 7) - (n / 7) + 6;
        int r1 = (i % 7) - (n % 7) + 6;
        s += btab[(r0 * 13 + r1) * NH_ + h];
        at[i][t] = s;
    }
    __syncthreads();

    // per-row: gate = sigmoid(mean), fused softmax * gate
    if (tid < N_) {
        float mx = -INFINITY, mean = 0.f;
        for (int t2 = 0; t2 < TS_; ++t2) {
            float v = at[tid][t2];
            mean += v; mx = fmaxf(mx, v);
        }
        mean *= (1.0f / TS_);
        float gate = 1.0f / (1.0f + expf(-mean));
        float se = 0.f;
        for (int t2 = 0; t2 < TS_; ++t2) se += expf(at[tid][t2] - mx);
        float sc = gate / se;
        for (int t2 = 0; t2 < TS_; ++t2) at[tid][t2] = expf(at[tid][t2] - mx) * sc;
    }
    __syncthreads();

    // o[i,d] -> (B,N,C) layout for the final projection GEMM
    for (int idx = tid; idx < N_ * HD_; idx += 64) {
        int i = idx / HD_, d = idx % HD_;
        float s = 0.f;
        for (int t2 = 0; t2 < TS_; ++t2) s += at[i][t2] * vh[t2][d];
        obuf[((size_t)(b * N_) + i) * C_ + h * HD_ + d] = s;
    }
}

// ======================================================================
// Host-side orchestration
// ======================================================================
extern "C" void kernel_launch(void* const* d_in, const int* in_sizes, int n_in,
                              void* d_out, int out_size, void* d_ws, size_t ws_size,
                              hipStream_t stream)
{
    const float* x   = (const float*)d_in[0];
    const float* Wq  = (const float*)d_in[1];
    const float* bq  = (const float*)d_in[2];
    const float* Wk  = (const float*)d_in[3];
    const float* bk  = (const float*)d_in[4];
    const float* Wv  = (const float*)d_in[5];
    const float* bv  = (const float*)d_in[6];
    const float* Wp  = (const float*)d_in[7];
    const float* bp  = (const float*)d_in[8];
    const float* Wc  = (const float*)d_in[9];
    const float* bc  = (const float*)d_in[10];
    const float* bt  = (const float*)d_in[11];

    // workspace layout (all sizes multiples of 4 floats -> 16B alignment kept)
    constexpr size_t NQ   = (size_t)B_ * N_ * C_;        // 77,070,336
    constexpr size_t NIDX = (size_t)B_ * NH_ * 32;       // 786,432
    constexpr size_t NQC  = (size_t)B_ * N_ * C2_;       // 38,535,168
    constexpr size_t NV   = (size_t)B_ * TS_ * C_;       // 50,331,648

    char* base = (char*)d_ws;
    float* q     = (float*)(base);
    int*   cidx  = (int*)  (base + NQ * 4);
    int*   pidx  = (int*)  (base + (NQ + NIDX) * 4);
    float* qcha1 = (float*)(base + (NQ + 2 * NIDX) * 4);
    float* kbuf  = (float*)(base + (NQ + 2 * NIDX + NQC) * 4);
    float* vin   = (float*)(base + (NQ + 2 * NIDX + 2 * NQC) * 4);
    float* vbuf  = (float*)(base + (NQ + 2 * NIDX + 2 * NQC + NV) * 4);
    float* obuf  = (float*)(base + (NQ + 2 * NIDX + 2 * NQC + 2 * NV) * 4);
    float* out   = (float*)d_out;

    const int M  = B_ * N_;      // 100352 = 784*128
    const int Mv = B_ * TS_;     // 65536  = 512*128

    // 1) q = x * Wq^T + bq
    gemm_xwt_bias<<<dim3(C_ / 128, M / 128), 256, 0, stream>>>(x, Wq, bq, q, M, C_, C_);

    // 2) channel + token top-k selection per (b,h)
    pool_select_kernel<<<dim3(B_ * NH_), 64, 0, stream>>>(q, Wc, bc, cidx, pidx);

    // 3) gather channel-selected q -> (B*N, C2)
    gather_qcha1<<<dim3((unsigned)(NQC / 256)), 256, 0, stream>>>(q, cidx, qcha1);

    // 4) k = qcha1 * Wk^T + bk
    gemm_xwt_bias<<<dim3(C2_ / 128, M / 128), 256, 0, stream>>>(qcha1, Wk, bk, kbuf, M, C2_, C2_);

    // 5) gather token-selected q -> (B*TS, C)
    gather_vin<<<dim3((unsigned)(NV / 256)), 256, 0, stream>>>(q, pidx, vin);

    // 6) v = vin * Wv^T + bv
    gemm_xwt_bias<<<dim3(C_ / 128, Mv / 128), 256, 0, stream>>>(vin, Wv, bv, vbuf, Mv, C_, C_);

    // 7) attention per (b,h) -> obuf in (B,N,C)
    attn_kernel<<<dim3(B_ * NH_), 64, 0, stream>>>(q, kbuf, vbuf, bt, cidx, pidx, obuf);

    // 8) out = obuf * Wp^T + bp
    gemm_xwt_bias<<<dim3(C_ / 128, M / 128), 256, 0, stream>>>(obuf, Wp, bp, out, M, C_, C_);
}
